// ChessGraphPooling_17506286698852
// MI455X (gfx1250) — matrix operand order repaired and checked by reading, not compiled, not run on hardware
//
#include <hip/hip_runtime.h>
#include <hip/hip_bf16.h>
#include <stdint.h>

// ---------------------------------------------------------------------------
// Types for CDNA5 WMMA (wave32): v_wmma_f32_16x16x32_bf16
// ---------------------------------------------------------------------------
typedef __attribute__((ext_vector_type(16))) __bf16 v16bf;
typedef __attribute__((ext_vector_type(8)))  __bf16 v8bf;
typedef __attribute__((ext_vector_type(8)))  float  v8f;

#define CC 256
#define NODES 64
#define BB 4096
#define NN (BB * NODES)   // 262144 nodes

// ---------------------------------------------------------------------------
// Async stage of a 16-row A stripe [16 x K] bf16 into LDS, shared by the
// block's 4 waves. Uses CDNA5 GLOBAL_LOAD_ASYNC_TO_LDS_B128 (ASYNCcnt).
// 16*K*2 bytes; K multiple of 64 -> chunk count multiple of 128 (no tail).
// ---------------------------------------------------------------------------
__device__ __forceinline__ void stage_a_async(const __bf16* __restrict__ A,
                                              int row0, int K,
                                              __bf16* __restrict__ smem) {
  const int tid = threadIdx.x;               // 128 threads
  const int rowBytes = K * 2;
  const uint32_t lbase = (uint32_t)(uintptr_t)smem;   // low 32 bits = LDS offset
  const int chunks = (16 * rowBytes) >> 4;   // 16-byte chunks
  for (int c = tid; c < chunks; c += 128) {
    int off = c << 4;
    int r = off / rowBytes;
    int inrow = off - r * rowBytes;
    uint64_t gaddr = (uint64_t)(uintptr_t)A + (uint64_t)(row0 + r) * rowBytes + inrow;
    uint32_t laddr = lbase + off;
    asm volatile("global_load_async_to_lds_b128 %0, %1, off"
                 :: "v"(laddr), "v"(gaddr) : "memory");
  }
  asm volatile("s_wait_asynccnt 0x0" ::: "memory");
  __syncthreads();
}

// Load A fragment for one 16x32 K-slab from the LDS stripe.
// A k order used here (B matches): lanes 0-15 hold K {0..15}, lanes 16-31 {16..31}.
__device__ __forceinline__ v16bf a_frag_lds(const __bf16* __restrict__ smem,
                                            int K, int lane15, int laneHi, int kk) {
  const __bf16* p = smem + (size_t)lane15 * K + (laneHi ? 16 : 0) + kk;
  union { v16bf v; v8bf h[2]; } ua;
  ua.h[0] = *(const v8bf*)(p);        // ds_load_b128
  ua.h[1] = *(const v8bf*)(p + 8);
  return ua.v;
}

// ---------------------------------------------------------------------------
// Weight packing: dst[n*K + k] = bf16(src[k*Nc + n])   (transpose + convert)
// ---------------------------------------------------------------------------
__global__ void pack_wt(const float* __restrict__ src, __bf16* __restrict__ dst,
                        int K, int Nc) {
  int idx = blockIdx.x * blockDim.x + threadIdx.x;
  if (idx < K * Nc) {
    int n = idx / K;
    int k = idx - n * K;
    dst[idx] = (__bf16)src[(size_t)k * Nc + n];
  }
}

// Fused node-level weights: 768 cols = [att(512) | piece(128) | empty(128)]
__global__ void pack_node_w(const float* __restrict__ attW1,
                            const float* __restrict__ pieceW1,
                            const float* __restrict__ emptyW1,
                            __bf16* __restrict__ dst) {
  int idx = blockIdx.x * blockDim.x + threadIdx.x;   // 768*256
  if (idx >= 768 * CC) return;
  int j = idx / CC;
  int k = idx - j * CC;
  float v;
  if (j < 512) {
    int h = j >> 6, d = j & 63;
    v = attW1[((size_t)h * CC + k) * 64 + d];
  } else if (j < 640) {
    v = pieceW1[(size_t)k * 128 + (j - 512)];
  } else {
    v = emptyW1[(size_t)k * 128 + (j - 640)];
  }
  dst[(size_t)j * CC + k] = (__bf16)v;
}

// bias_nd[768] = [att_b1 | piece_b1 | empty_b1];  w2cat[768] = [att_w2 | piece_w2 | empty_w2]
__global__ void pack_node_vec(const float* __restrict__ ab1, const float* __restrict__ pb1,
                              const float* __restrict__ eb1, const float* __restrict__ aw2,
                              const float* __restrict__ pw2, const float* __restrict__ ew2,
                              float* __restrict__ bias_nd, float* __restrict__ w2cat) {
  int j = blockIdx.x * blockDim.x + threadIdx.x;
  if (j >= 768) return;
  if (j < 512)      { bias_nd[j] = ab1[j];       w2cat[j] = aw2[j]; }
  else if (j < 640) { bias_nd[j] = pb1[j - 512]; w2cat[j] = pw2[j - 512]; }
  else              { bias_nd[j] = eb1[j - 640]; w2cat[j] = ew2[j - 640]; }
}

// ---------------------------------------------------------------------------
// x -> bf16, and x * sigmoid(strat_w[square]) -> bf16
// ---------------------------------------------------------------------------
__global__ void convert_x(const float* __restrict__ x,
                          const float* __restrict__ strat_w,
                          __bf16* __restrict__ xb, __bf16* __restrict__ xs,
                          size_t total) {
  size_t i = (size_t)blockIdx.x * blockDim.x + threadIdx.x;
  if (i >= total) return;
  float v = x[i];
  xb[i] = (__bf16)v;
  int node = (int)(i >> 8);
  int sq = node & (NODES - 1);
  float sa = 1.0f / (1.0f + __expf(-strat_w[sq]));
  xs[i] = (__bf16)(v * sa);
}

// ---------------------------------------------------------------------------
// Generic bf16 WMMA GEMM: C = act(A[M,K] @ Bt^T + bias).
// Block = 128 threads (4 waves); each wave owns 16 rows x 64 cols (4 WMMA
// tiles, one A fragment amortized over 4). A stripe staged in LDS (async).
// Requires: K % 32 == 0, M % 16 == 0, Nc == gridDim.y * 256.
// act: 0 = none, 1 = LeakyReLU(0.2). Dyn LDS = 32*K bytes.
// ---------------------------------------------------------------------------
__global__ void __launch_bounds__(128)
wmma_gemm4(const __bf16* __restrict__ A, const __bf16* __restrict__ Bt,
           const float* __restrict__ bias, int M, int K,
           float* __restrict__ Cf, __bf16* __restrict__ Cb,
           int ldc, int col_off, int act) {
  extern __shared__ __bf16 smem[];
  const int lane   = threadIdx.x & 31;
  const int wave   = threadIdx.x >> 5;
  const int row0   = blockIdx.x * 16;
  const int colG   = (blockIdx.y * 4 + wave) * 64;
  const int laneHi = lane >> 4;
  const int lane15 = lane & 15;

  stage_a_async(A, row0, K, smem);

  const __bf16* brow0 = Bt + (size_t)(colG + lane15) * K + (laneHi ? 16 : 0);

  v8f acc[4];
#pragma unroll
  for (int t = 0; t < 4; ++t) acc[t] = v8f{0.f,0.f,0.f,0.f,0.f,0.f,0.f,0.f};

  for (int kk = 0; kk < K; kk += 32) {
    v16bf a = a_frag_lds(smem, K, lane15, laneHi, kk);
    if (kk + 32 < K) __builtin_prefetch(brow0 + kk + 32, 0, 3);
#pragma unroll
    for (int t = 0; t < 4; ++t) {
      v16bf b = *(const v16bf*)(brow0 + (size_t)t * 16 * K + kk);
      acc[t] = __builtin_amdgcn_wmma_f32_16x16x32_bf16(
          false, a, false, b, (short)0, acc[t], false, false);
    }
  }

#pragma unroll
  for (int t = 0; t < 4; ++t) {
    int cg = colG + t * 16 + lane15;
    float bv = bias ? bias[cg] : 0.f;
#pragma unroll
    for (int v = 0; v < 8; ++v) {
      int m = row0 + v + laneHi * 8;
      float val = acc[t][v] + bv;
      if (act == 1) val = (val > 0.f) ? val : 0.2f * val;
      size_t o = (size_t)m * ldc + col_off + cg;
      if (Cf) Cf[o] = val;
      else    Cb[o] = (__bf16)val;
    }
  }
}

// ---------------------------------------------------------------------------
// Fused node GEMM + score reduction. Computes
//   hidden = LeakyReLU(x @ Wnode + b1)  (never materialized)
//   S[n, ch] += sum_cols hidden * w2cat   (ch: 0-7 att heads, 8 piece, 9 empty)
// Each wave owns one 64-col group == exactly one score channel segment slice.
// Partial row sums combined with atomicAdd into pre-zeroed S[N,10].
// ---------------------------------------------------------------------------
__global__ void __launch_bounds__(128)
wmma_node_score(const __bf16* __restrict__ A, const __bf16* __restrict__ Bt,
                const float* __restrict__ bias, const float* __restrict__ w2cat,
                float* __restrict__ S) {
  extern __shared__ __bf16 smem[];
  const int K = CC;
  const int lane   = threadIdx.x & 31;
  const int wave   = threadIdx.x >> 5;
  const int row0   = blockIdx.x * 16;
  const int colG   = (blockIdx.y * 4 + wave) * 64;   // gridDim.y = 3 -> 0..704
  const int laneHi = lane >> 4;
  const int lane15 = lane & 15;
  const int ch = (colG < 512) ? (colG >> 6) : ((colG < 640) ? 8 : 9);

  stage_a_async(A, row0, K, smem);

  const __bf16* brow0 = Bt + (size_t)(colG + lane15) * K + (laneHi ? 16 : 0);

  v8f acc[4];
#pragma unroll
  for (int t = 0; t < 4; ++t) acc[t] = v8f{0.f,0.f,0.f,0.f,0.f,0.f,0.f,0.f};

  for (int kk = 0; kk < K; kk += 32) {
    v16bf a = a_frag_lds(smem, K, lane15, laneHi, kk);
    if (kk + 32 < K) __builtin_prefetch(brow0 + kk + 32, 0, 3);
#pragma unroll
    for (int t = 0; t < 4; ++t) {
      v16bf b = *(const v16bf*)(brow0 + (size_t)t * 16 * K + kk);
      acc[t] = __builtin_amdgcn_wmma_f32_16x16x32_bf16(
          false, a, false, b, (short)0, acc[t], false, false);
    }
  }

  float bv[4], wv[4];
#pragma unroll
  for (int t = 0; t < 4; ++t) {
    int cg = colG + t * 16 + lane15;
    bv[t] = bias[cg];
    wv[t] = w2cat[cg];
  }
#pragma unroll
  for (int v = 0; v < 8; ++v) {
    float contrib = 0.f;
#pragma unroll
    for (int t = 0; t < 4; ++t) {
      float val = acc[t][v] + bv[t];
      val = (val > 0.f) ? val : 0.2f * val;      // LeakyReLU(0.2)
      contrib += val * wv[t];
    }
    // reduce over the 16 lanes of this half-wave (rows 0-7 in lanes 0-15,
    // rows 8-15 in lanes 16-31, matching the f32 C/D fragment layout)
#pragma unroll
    for (int o = 8; o > 0; o >>= 1) contrib += __shfl_xor(contrib, o, 16);
    if (lane15 == 0) {
      int m = row0 + v + laneHi * 8;
      atomicAdd(&S[(size_t)m * 10 + ch], contrib);
    }
  }
}

// ---------------------------------------------------------------------------
// Per-graph segment softmax (10 channels over 64 nodes) + weighted pooling.
// Applies w2-stage biases and piece/empty masks at load time.
// ---------------------------------------------------------------------------
__global__ void __launch_bounds__(256)
softmax_pool(const float* __restrict__ S, const float* __restrict__ x,
             const float* __restrict__ att_b2, const float* __restrict__ piece_b2,
             const float* __restrict__ empty_b2, const int* __restrict__ node_types,
             __bf16* __restrict__ att_cat, __bf16* __restrict__ pp_ep) {
  const int b = blockIdx.x;
  const int tid = threadIdx.x;
  __shared__ float w[NODES][10];

  for (int i = tid; i < NODES * 10; i += 256) {
    int n = i / 10, ch = i - n * 10;
    int gn = b * NODES + n;
    float raw = S[(size_t)gn * 10 + ch];
    float v;
    if (ch < 8) {
      v = raw + att_b2[ch];
    } else {
      float mask = (node_types[gn] > 0) ? 1.f : 0.f;
      if (ch == 8) v = (raw + piece_b2[0]) * mask;
      else         v = (raw + empty_b2[0]) * (1.f - mask);
    }
    w[n][ch] = v;
  }
  __syncthreads();

  if (tid < 10) {
    float m = -3.402823466e+38f;
    for (int n = 0; n < NODES; ++n) m = fmaxf(m, w[n][tid]);
    float s = 0.f;
    for (int n = 0; n < NODES; ++n) {
      float e = __expf(w[n][tid] - m);
      w[n][tid] = e;
      s += e;
    }
    float inv = 1.f / (s + 1e-16f);
    for (int n = 0; n < NODES; ++n) w[n][tid] *= inv;
  }
  __syncthreads();

  float acc[10];
#pragma unroll
  for (int ch = 0; ch < 10; ++ch) acc[ch] = 0.f;
  const float* xb = x + (size_t)b * NODES * CC;
  for (int n = 0; n < NODES; ++n) {
    float xv = xb[(size_t)n * CC + tid];
#pragma unroll
    for (int ch = 0; ch < 10; ++ch) acc[ch] += xv * w[n][ch];
  }
#pragma unroll
  for (int i = 0; i < 8; ++i)
    att_cat[(size_t)b * 2048 + i * CC + tid] = (__bf16)acc[i];
  pp_ep[(size_t)b * 512 + tid]      = (__bf16)acc[8];
  pp_ep[(size_t)b * 512 + CC + tid] = (__bf16)acc[9];
}

// ---------------------------------------------------------------------------
// Row LayerNorm + ReLU. 256 threads per row. f32 (outF) or bf16 (outB) out.
// ---------------------------------------------------------------------------
__global__ void __launch_bounds__(256)
ln_relu(const float* __restrict__ X, const float* __restrict__ g,
        const float* __restrict__ beta, int NC,
        float* __restrict__ outF, __bf16* __restrict__ outB,
        int ldo, int col_off) {
  const int row = blockIdx.x;
  const int tid = threadIdx.x;
  const float* xr = X + (size_t)row * NC;
  float s = 0.f, s2 = 0.f;
  for (int j = tid; j < NC; j += 256) {
    float v = xr[j];
    s += v;
    s2 += v * v;
  }
  __shared__ float r1[256], r2[256];
  r1[tid] = s; r2[tid] = s2;
  __syncthreads();
  for (int off = 128; off > 0; off >>= 1) {
    if (tid < off) { r1[tid] += r1[tid + off]; r2[tid] += r2[tid + off]; }
    __syncthreads();
  }
  float mu = r1[0] / NC;
  float var = r2[0] / NC - mu * mu;
  float rstd = rsqrtf(var + 1e-5f);
  for (int j = tid; j < NC; j += 256) {
    float v = (xr[j] - mu) * rstd * g[j] + beta[j];
    v = (v > 0.f) ? v : 0.f;
    size_t o = (size_t)row * ldo + col_off + j;
    if (outF) outF[o] = v;
    else      outB[o] = (__bf16)v;
  }
}

// ---------------------------------------------------------------------------
// Strategic mean-pool: sf[N,256] f32 -> combined[:,512:768] bf16 (mean of 64)
// ---------------------------------------------------------------------------
__global__ void __launch_bounds__(256)
pool_s(const float* __restrict__ sf, __bf16* __restrict__ combined) {
  const int b = blockIdx.x;
  const int c = threadIdx.x;
  const float* p = sf + (size_t)b * NODES * CC + c;
  float s = 0.f;
#pragma unroll 8
  for (int n = 0; n < NODES; ++n) s += p[(size_t)n * CC];
  combined[(size_t)b * 768 + 512 + c] = (__bf16)(s * (1.f / 64.f));
}

// ===========================================================================
extern "C" void kernel_launch(void* const* d_in, const int* in_sizes, int n_in,
                              void* d_out, int out_size, void* d_ws, size_t ws_size,
                              hipStream_t stream) {
  (void)in_sizes; (void)n_in; (void)out_size; (void)ws_size;
  const float* x        = (const float*)d_in[0];
  const float* att_W1   = (const float*)d_in[1];
  const float* att_b1   = (const float*)d_in[2];
  const float* att_w2   = (const float*)d_in[3];
  const float* att_b2   = (const float*)d_in[4];
  const float* comb_W   = (const float*)d_in[5];
  const float* comb_b   = (const float*)d_in[6];
  const float* comb_g   = (const float*)d_in[7];
  const float* comb_be  = (const float*)d_in[8];
  const float* piece_W1 = (const float*)d_in[9];
  const float* piece_b1 = (const float*)d_in[10];
  const float* piece_w2 = (const float*)d_in[11];
  const float* piece_b2 = (const float*)d_in[12];
  const float* empty_W1 = (const float*)d_in[13];
  const float* empty_b1 = (const float*)d_in[14];
  const float* empty_w2 = (const float*)d_in[15];
  const float* empty_b2 = (const float*)d_in[16];
  const float* hier_W   = (const float*)d_in[17];
  const float* hier_b   = (const float*)d_in[18];
  const float* hier_g   = (const float*)d_in[19];
  const float* hier_be  = (const float*)d_in[20];
  const float* strat_w  = (const float*)d_in[21];
  const float* sp_W     = (const float*)d_in[22];
  const float* sp_b     = (const float*)d_in[23];
  const float* sp_g     = (const float*)d_in[24];
  const float* sp_be    = (const float*)d_in[25];
  const float* p1_W     = (const float*)d_in[26];
  const float* p1_b     = (const float*)d_in[27];
  const float* p1_g     = (const float*)d_in[28];
  const float* p1_be    = (const float*)d_in[29];
  const float* p2_W     = (const float*)d_in[30];
  const float* p2_b     = (const float*)d_in[31];
  const int*   node_types = (const int*)d_in[33];
  float* out = (float*)d_out;   // [4096, 256] f32

  // ---- workspace carve-out (256B aligned) ----
  char* ws = (char*)d_ws;
  size_t cur = 0;
  auto alloc = [&](size_t bytes) -> void* {
    void* p = ws + cur;
    cur = (cur + bytes + 255) & ~(size_t)255;
    return p;
  };
  __bf16* xb       = (__bf16*)alloc((size_t)NN * CC * 2);
  __bf16* xs       = (__bf16*)alloc((size_t)NN * CC * 2);
  float*  sf       = (float*) alloc((size_t)NN * CC * 4);
  float*  S        = (float*) alloc((size_t)NN * 10 * 4);
  __bf16* att_cat  = (__bf16*)alloc((size_t)BB * 2048 * 2);
  __bf16* pp_ep    = (__bf16*)alloc((size_t)BB * 512 * 2);
  __bf16* combined = (__bf16*)alloc((size_t)BB * 768 * 2);
  __bf16* z        = (__bf16*)alloc((size_t)BB * 512 * 2);
  float*  tmp      = (float*) alloc((size_t)BB * 512 * 4);
  __bf16* Wnode_t  = (__bf16*)alloc((size_t)768 * CC * 2);
  float*  bias_nd  = (float*) alloc(768 * 4);
  float*  w2cat    = (float*) alloc(768 * 4);
  __bf16* spWt     = (__bf16*)alloc((size_t)CC * CC * 2);
  __bf16* combWt   = (__bf16*)alloc((size_t)CC * 2048 * 2);
  __bf16* hierWt   = (__bf16*)alloc((size_t)CC * 512 * 2);
  __bf16* p1Wt     = (__bf16*)alloc((size_t)512 * 768 * 2);
  __bf16* p2Wt     = (__bf16*)alloc((size_t)CC * 512 * 2);

  // ---- 0. zero score accumulator (graph-capturable async memset) ----
  hipMemsetAsync(S, 0, (size_t)NN * 10 * 4, stream);

  // ---- 1. pack weights (small) ----
  pack_node_w<<<(768 * CC + 255) / 256, 256, 0, stream>>>(att_W1, piece_W1, empty_W1, Wnode_t);
  pack_node_vec<<<3, 256, 0, stream>>>(att_b1, piece_b1, empty_b1,
                                       att_w2, piece_w2, empty_w2, bias_nd, w2cat);
  pack_wt<<<(CC * CC + 255) / 256, 256, 0, stream>>>(sp_W, spWt, CC, CC);
  pack_wt<<<(2048 * CC + 255) / 256, 256, 0, stream>>>(comb_W, combWt, 2048, CC);
  pack_wt<<<(512 * CC + 255) / 256, 256, 0, stream>>>(hier_W, hierWt, 512, CC);
  pack_wt<<<(768 * 512 + 255) / 256, 256, 0, stream>>>(p1_W, p1Wt, 768, 512);
  pack_wt<<<(512 * CC + 255) / 256, 256, 0, stream>>>(p2_W, p2Wt, 512, CC);

  // ---- 2. convert x ----
  {
    size_t total = (size_t)NN * CC;
    convert_x<<<(unsigned)((total + 255) / 256), 256, 0, stream>>>(x, strat_w, xb, xs, total);
  }

  // ---- 3. fused node GEMM + score reduction (no hidden materialization) ----
  wmma_node_score<<<dim3(NN / 16, 3), 128, 32 * CC, stream>>>(
      xb, Wnode_t, bias_nd, w2cat, S);

  // ---- 4. strategic GEMM + node-level LN/ReLU (in place) + mean pool ----
  wmma_gemm4<<<dim3(NN / 16, 1), 128, 32 * CC, stream>>>(
      xs, spWt, sp_b, NN, CC, sf, nullptr, CC, 0, 0);
  ln_relu<<<NN, 256, 0, stream>>>(sf, sp_g, sp_be, CC, sf, nullptr, CC, 0);
  pool_s<<<BB, 256, 0, stream>>>(sf, combined);          // combined[:,512:768]

  // ---- 5. segment softmax + attention / hierarchical pooling ----
  softmax_pool<<<BB, 256, 0, stream>>>(S, x, att_b2, piece_b2, empty_b2,
                                       node_types, att_cat, pp_ep);

  // ---- 6. pooled_att = ReLU(LN(att_cat @ comb_W + b)) -> combined[:,0:256] ----
  wmma_gemm4<<<dim3(BB / 16, 1), 128, 32 * 2048, stream>>>(
      att_cat, combWt, comb_b, BB, 2048, tmp, nullptr, CC, 0, 0);
  ln_relu<<<BB, 256, 0, stream>>>(tmp, comb_g, comb_be, CC, nullptr, combined, 768, 0);

  // ---- 7. pooled_h = ReLU(LN(pp_ep @ hier_W + b)) -> combined[:,256:512] ----
  wmma_gemm4<<<dim3(BB / 16, 1), 128, 32 * 512, stream>>>(
      pp_ep, hierWt, hier_b, BB, 512, tmp, nullptr, CC, 0, 0);
  ln_relu<<<BB, 256, 0, stream>>>(tmp, hier_g, hier_be, CC, nullptr, combined, 768, 256);

  // ---- 8. z = ReLU(LN(combined @ p1_W + b)), [4096,512] ----
  wmma_gemm4<<<dim3(BB / 16, 2), 128, 32 * 768, stream>>>(
      combined, p1Wt, p1_b, BB, 768, tmp, nullptr, 512, 0, 0);
  ln_relu<<<BB, 256, 0, stream>>>(tmp, p1_g, p1_be, 512, nullptr, z, 512, 0);

  // ---- 9. out = z @ p2_W + b, [4096,256] f32 straight to d_out ----
  wmma_gemm4<<<dim3(BB / 16, 1), 128, 32 * 512, stream>>>(
      z, p2Wt, p2_b, BB, 512, out, nullptr, CC, 0, 0);
}
